// InterpolationModGeoEC_68126771249260
// MI455X (gfx1250) — matrix operand order repaired and checked by reading, not compiled, not run on hardware
//
#include <hip/hip_runtime.h>

#define FWID    648
#define HID     128
#define SIGMA_C 100.0f

typedef __attribute__((ext_vector_type(16))) _Float16 v16h;
typedef __attribute__((ext_vector_type(8)))  _Float16 v8h;
typedef __attribute__((ext_vector_type(8)))  float    v8f;

#if defined(__gfx1250__) && __has_builtin(__builtin_amdgcn_global_load_async_to_lds_b32)
#define HAVE_ASYNC_LDS 1
#endif
#define AS1 __attribute__((address_space(1)))
#define AS3 __attribute__((address_space(3)))

// ---------------------------------------------------------------------------
// WMMA fragment helpers (CDNA5 wave32 layouts, CDNA5 ISA §7.12.2)
// A 16x32 f16: lane L -> row L&15; halves 0..7 at k=kb..kb+7, 8..15 at k=kb+16..kb+23, kb=(L>>4)*8
// B 32x16 f16 from BT[N x Kpad] (BT[n][k]=W[k][n]): lane L -> col L&15, k order same as A
// C/D f32 16x16: vgpr v -> row v+8*(L>>4), col L&15
// ---------------------------------------------------------------------------
__device__ __forceinline__ v16h make_a_frag(const _Float16* p) {
  v8h lo = *(const v8h*)p;
  v8h hi = *(const v8h*)(p + 16);
  return __builtin_shufflevector(lo, hi, 0,1,2,3,4,5,6,7,8,9,10,11,12,13,14,15);
}

// A fragment from an f32 staging tile (row base + kb already applied)
__device__ __forceinline__ v16h a_frag_from_f32(const float* p) {
  v16h a;
#pragma unroll
  for (int h = 0; h < 8; ++h) { a[h] = (_Float16)p[h]; a[h + 8] = (_Float16)p[h + 16]; }
  return a;
}

__device__ __forceinline__ void atomicMaxF(float* a, float v) {
  int old = __float_as_int(*a);
  while (__int_as_float(old) < v) {
    int prev = atomicCAS((int*)a, old, __float_as_int(v));
    if (prev == old) break;
    old = prev;
  }
}

// ---------------------------------------------------------------------------
// small utility kernels
// ---------------------------------------------------------------------------
__global__ void k_fill(float* p, float v, int n) {
  int i = blockIdx.x * blockDim.x + threadIdx.x;
  if (i < n) p[i] = v;
}

__global__ void k_count(const int* __restrict__ dst, int E, float* __restrict__ cnt) {
  int e = blockIdx.x * blockDim.x + threadIdx.x;
  if (e < E) atomicAdd(&cnt[dst[e]], 1.0f);
}

__global__ void k_init_feat(const float* __restrict__ v, const float* __restrict__ nrm,
                            float* __restrict__ F, int N) {
  int i = blockIdx.x * blockDim.x + threadIdx.x;
  if (i >= N) return;
  float* r = F + (size_t)i * FWID + 640;
  r[0] = v[i*3+0]; r[1] = v[i*3+1]; r[2] = v[i*3+2];
  r[3] = nrm[i*3+0]; r[4] = nrm[i*3+1]; r[5] = nrm[i*3+2];
}

__global__ void k_init_rn(const float* __restrict__ vx, const float* __restrict__ mapped,
                          float t, float* __restrict__ F, int N) {
  int i = blockIdx.x * blockDim.x + threadIdx.x;
  if (i >= N) return;
  float* r = F + (size_t)i * FWID + 640;
  r[0] = vx[i*3+0]; r[1] = vx[i*3+1]; r[2] = vx[i*3+2];
  r[3] = mapped[i*3+0] - vx[i*3+0];
  r[4] = mapped[i*3+1] - vx[i*3+1];
  r[5] = mapped[i*3+2] - vx[i*3+2];
  r[6] = t;
}

// W [K x N] f32 row-major  ->  BT [N x Kpad] f16 (zero-padded K)
__global__ void k_wt_trans(const float* __restrict__ W, _Float16* __restrict__ BT,
                           int K, int Ncols, int Kpad) {
  int idx = blockIdx.x * blockDim.x + threadIdx.x;
  if (idx >= Ncols * Kpad) return;
  int n = idx / Kpad, k = idx % Kpad;
  BT[idx] = (k < K) ? (_Float16)W[(size_t)k * Ncols + n] : (_Float16)0.0f;
}

// ---------------------------------------------------------------------------
// EdgeConv GEMM1: H1[e,:] = relu( [F[dst], F[src]-F[dst]] @ W1 + b1 )  (f16 out)
// 32-edge x 128-col block; 8 waves; each wave: 2 M-subtiles, B frag reused.
// ---------------------------------------------------------------------------
__global__ __launch_bounds__(256) void k_ec_gemm1(
    const float* __restrict__ F, int colOfs, int w,
    const int* __restrict__ src, const int* __restrict__ dst, int E,
    const _Float16* __restrict__ BT, int Kp, const float* __restrict__ bias,
    _Float16* __restrict__ H1) {
  __shared__ _Float16 sA[32 * 32];
  __shared__ int sDst[32], sSrc[32];
  const int e0   = blockIdx.x * 32;
  const int tid  = threadIdx.x;
  const int lane = tid & 31;
  const int wv   = tid >> 5;
  const int col  = wv * 16 + (lane & 15);
  const int hi16 = lane >> 4;
  if (tid < 32) {
    int e = e0 + tid; if (e > E - 1) e = E - 1;
    sDst[tid] = dst[e]; sSrc[tid] = src[e];
  }
  v8f acc0 = {}, acc1 = {};
  const int nkt = Kp >> 5;
  for (int kt = 0; kt < nkt; ++kt) {
    __syncthreads();
    for (int s = tid; s < 1024; s += 256) {
      const int row = s >> 5;
      const int k   = (kt << 5) + (s & 31);
      float val = 0.0f;
      if (k < 2 * w) {
        const int di = sDst[row];
        if (k < w) {
          val = F[(size_t)di * FWID + colOfs + k];
        } else {
          const int si = sSrc[row];
          const int kc = k - w;
          val = F[(size_t)si * FWID + colOfs + kc] - F[(size_t)di * FWID + colOfs + kc];
        }
      }
      sA[s] = (_Float16)val;
    }
    __syncthreads();
    v16h b = make_a_frag(BT + (size_t)col * Kp + (kt << 5) + hi16 * 8);
    if (kt + 1 < nkt)
      __builtin_prefetch(BT + (size_t)col * Kp + ((kt + 1) << 5) + hi16 * 16, 0, 3);
    v16h a0 = make_a_frag(sA + (lane & 15) * 32 + hi16 * 8);
    v16h a1 = make_a_frag(sA + (16 + (lane & 15)) * 32 + hi16 * 8);
    acc0 = __builtin_amdgcn_wmma_f32_16x16x32_f16(false, a0, false, b, (short)0, acc0, false, false);
    acc1 = __builtin_amdgcn_wmma_f32_16x16x32_f16(false, a1, false, b, (short)0, acc1, false, false);
  }
  const float bv = bias[col];
#pragma unroll
  for (int v = 0; v < 8; ++v) {
    const int r  = v + 8 * hi16;
    const int e0r = e0 + r;
    if (e0r < E)      H1[(size_t)e0r * HID + col]        = (_Float16)fmaxf(acc0[v] + bv, 0.0f);
    if (e0r + 16 < E) H1[(size_t)(e0r + 16) * HID + col] = (_Float16)fmaxf(acc1[v] + bv, 0.0f);
  }
}

// ---------------------------------------------------------------------------
// EdgeConv GEMM2 + scatter: S[dst[e],:] += H1[e,:] @ W2 + b2
// ---------------------------------------------------------------------------
__global__ __launch_bounds__(256) void k_ec_gemm2(
    const _Float16* __restrict__ H1, const int* __restrict__ dst, int E,
    const _Float16* __restrict__ BT, const float* __restrict__ bias,
    float* __restrict__ S) {
  const int e0   = blockIdx.x * 32;
  const int tid  = threadIdx.x;
  const int lane = tid & 31;
  const int wv   = tid >> 5;
  const int col  = wv * 16 + (lane & 15);
  const int hi16 = lane >> 4;
  int ra = e0 + (lane & 15);
  if (ra > E - 17) ra = E - 17;          // keep ra and ra+16 in range
  const _Float16* ar0 = H1 + (size_t)ra * HID;
  const _Float16* ar1 = ar0 + (size_t)16 * HID;
  v8f acc0 = {}, acc1 = {};
#pragma unroll
  for (int kt = 0; kt < 4; ++kt) {
    v16h b  = make_a_frag(BT + (size_t)col * HID + (kt << 5) + hi16 * 8);
    v16h a0 = make_a_frag(ar0 + (kt << 5) + hi16 * 8);
    v16h a1 = make_a_frag(ar1 + (kt << 5) + hi16 * 8);
    acc0 = __builtin_amdgcn_wmma_f32_16x16x32_f16(false, a0, false, b, (short)0, acc0, false, false);
    acc1 = __builtin_amdgcn_wmma_f32_16x16x32_f16(false, a1, false, b, (short)0, acc1, false, false);
  }
  const float bv = bias[col];
#pragma unroll
  for (int v = 0; v < 8; ++v) {
    const int e = e0 + v + 8 * hi16;
    if (e < E)      atomicAdd(&S[(size_t)dst[e] * HID + col],      acc0[v] + bv);
    if (e + 16 < E) atomicAdd(&S[(size_t)dst[e + 16] * HID + col], acc1[v] + bv);
  }
}

// segment mean into F columns, reset S
__global__ void k_mean_place(float* __restrict__ S, const float* __restrict__ cnt,
                             float* __restrict__ F, int dstCol, int N) {
  int idx = blockIdx.x * blockDim.x + threadIdx.x;
  if (idx >= N * HID) return;
  int i = idx >> 7, c = idx & 127;
  F[(size_t)i * FWID + dstCol + c] = S[idx] / fmaxf(cnt[i], 1.0f);
  S[idx] = 0.0f;
}

// ---------------------------------------------------------------------------
// Final linear (c_dim=128): out[N x 128] = F[:, :W] @ Wf + bf
// A tile staged f32 in LDS via async global->LDS copies when available.
// ---------------------------------------------------------------------------
__global__ __launch_bounds__(256) void k_fc_gemm(
    const float* __restrict__ F, int W, int N,
    const _Float16* __restrict__ BT, int Kp, const float* __restrict__ bias,
    float* __restrict__ out) {
  __shared__ float sAf[32 * 32];
  const int i0   = blockIdx.x * 32;
  const int tid  = threadIdx.x;
  const int lane = tid & 31;
  const int wv   = tid >> 5;
  const int col  = wv * 16 + (lane & 15);
  const int hi16 = lane >> 4;
  v8f acc0 = {}, acc1 = {};
  const int nkt = Kp >> 5;
  for (int kt = 0; kt < nkt; ++kt) {
    __syncthreads();
    for (int s = tid; s < 1024; s += 256) {
      const int row = s >> 5;
      const int k   = (kt << 5) + (s & 31);
      const int i   = i0 + row;
      if (i < N && k < W) {
#ifdef HAVE_ASYNC_LDS
        __builtin_amdgcn_global_load_async_to_lds_b32(
            (AS1 int*)(F + (size_t)i * FWID + k), (AS3 int*)&sAf[s], 0, 0);
#else
        sAf[s] = F[(size_t)i * FWID + k];
#endif
      } else {
        sAf[s] = 0.0f;
      }
    }
#ifdef HAVE_ASYNC_LDS
#if __has_builtin(__builtin_amdgcn_s_wait_asynccnt)
    __builtin_amdgcn_s_wait_asynccnt(0);
#else
    asm volatile("s_wait_asynccnt 0x0" ::: "memory");
#endif
#endif
    __syncthreads();
    v16h b  = make_a_frag(BT + (size_t)col * Kp + (kt << 5) + hi16 * 8);
    v16h a0 = a_frag_from_f32(sAf + (lane & 15) * 32 + hi16 * 8);
    v16h a1 = a_frag_from_f32(sAf + (16 + (lane & 15)) * 32 + hi16 * 8);
    acc0 = __builtin_amdgcn_wmma_f32_16x16x32_f16(false, a0, false, b, (short)0, acc0, false, false);
    acc1 = __builtin_amdgcn_wmma_f32_16x16x32_f16(false, a1, false, b, (short)0, acc1, false, false);
  }
  const float bv = bias[col];
#pragma unroll
  for (int v = 0; v < 8; ++v) {
    const int i = i0 + v + 8 * hi16;
    if (i < N)      out[(size_t)i * HID + col]        = acc0[v] + bv;
    if (i + 16 < N) out[(size_t)(i + 16) * HID + col] = acc1[v] + bv;
  }
}

// L2-normalize rows, emit f16
__global__ void k_normalize(const float* __restrict__ raw, _Float16* __restrict__ outh, int N) {
  int i = blockIdx.x * blockDim.x + threadIdx.x;
  if (i >= N) return;
  const float* r = raw + (size_t)i * HID;
  float s = 0.0f;
  for (int c = 0; c < HID; ++c) s += r[c] * r[c];
  float inv = rsqrtf(fmaxf(s, 1e-20f));
  for (int c = 0; c < HID; ++c) outh[(size_t)i * HID + c] = (_Float16)(r[c] * inv);
}

// ---------------------------------------------------------------------------
// D = fxh @ fyh^T passes (never materialized). One wave = 16x64 strip:
// A frag loaded once per K step, reused across 4 B tiles (16 WMMA / wave).
// mode 0: row/col max.  mode 1: rowsum, rowacc, colsum.  mode 2: colacc.
// ---------------------------------------------------------------------------
__global__ __launch_bounds__(256) void k_dpass(
    const _Float16* __restrict__ fxh, const _Float16* __restrict__ fyh,
    int Nx, int Ny, int mode,
    float* __restrict__ rowmax, float* __restrict__ colmax,
    float* __restrict__ rowsum, float* __restrict__ colsum,
    float* __restrict__ rowacc, const float* __restrict__ vy,
    float* __restrict__ colacc, const float* __restrict__ predl) {
  const int tid  = threadIdx.x;
  const int lane = tid & 31;
  const int wv   = tid >> 5;
  const int i0   = blockIdx.x * 16;
  const int j0   = (blockIdx.y * 8 + wv) * 64;
  if (j0 >= Ny) return;                        // wave-uniform exit (EXEC stays full)
  const int hi16 = lane >> 4;
  const int col  = lane & 15;
  const _Float16* arow = fxh + (size_t)(i0 + col) * HID;
  const _Float16* brow[4];
#pragma unroll
  for (int u = 0; u < 4; ++u) brow[u] = fyh + (size_t)(j0 + u * 16 + col) * HID;
  v8f acc[4] = {};
#pragma unroll
  for (int kt = 0; kt < 4; ++kt) {
    v16h a = make_a_frag(arow + (kt << 5) + hi16 * 8);
#pragma unroll
    for (int u = 0; u < 4; ++u) {
      v16h b = make_a_frag(brow[u] + (kt << 5) + hi16 * 8);
      acc[u] = __builtin_amdgcn_wmma_f32_16x16x32_f16(false, a, false, b, (short)0, acc[u], false, false);
    }
  }
  if (mode == 0) {
#pragma unroll
    for (int v = 0; v < 8; ++v) {
      float m = acc[0][v];
#pragma unroll
      for (int u = 1; u < 4; ++u) m = fmaxf(m, acc[u][v]);
      for (int o = 1; o < 16; o <<= 1) m = fmaxf(m, __shfl_xor(m, o));
      if (col == 0) atomicMaxF(&rowmax[i0 + v + 8 * hi16], m);
    }
#pragma unroll
    for (int u = 0; u < 4; ++u) {
      float cm = acc[u][0];
#pragma unroll
      for (int v = 1; v < 8; ++v) cm = fmaxf(cm, acc[u][v]);
      cm = fmaxf(cm, __shfl_xor(cm, 16));
      if (lane < 16) atomicMaxF(&colmax[j0 + u * 16 + lane], cm);
    }
  } else if (mode == 1) {
    float vyv[4][3], cmx[4], csum[4];
#pragma unroll
    for (int u = 0; u < 4; ++u) {
      const int j = j0 + u * 16 + col;
      vyv[u][0] = vy[j*3]; vyv[u][1] = vy[j*3+1]; vyv[u][2] = vy[j*3+2];
      cmx[u] = colmax[j]; csum[u] = 0.0f;
    }
#pragma unroll
    for (int v = 0; v < 8; ++v) {
      const int i = i0 + v + 8 * hi16;
      const float rmx = rowmax[i];
      float s = 0.0f, a0 = 0.0f, a1 = 0.0f, a2 = 0.0f;
#pragma unroll
      for (int u = 0; u < 4; ++u) {
        float ex = expf(SIGMA_C * (acc[u][v] - rmx));
        csum[u] += expf(SIGMA_C * (acc[u][v] - cmx[u]));
        s += ex; a0 += ex * vyv[u][0]; a1 += ex * vyv[u][1]; a2 += ex * vyv[u][2];
      }
      for (int o = 1; o < 16; o <<= 1) {
        s  += __shfl_xor(s, o);  a0 += __shfl_xor(a0, o);
        a1 += __shfl_xor(a1, o); a2 += __shfl_xor(a2, o);
      }
      if (col == 0) {
        atomicAdd(&rowsum[i], s);
        atomicAdd(&rowacc[i*3+0], a0);
        atomicAdd(&rowacc[i*3+1], a1);
        atomicAdd(&rowacc[i*3+2], a2);
      }
    }
#pragma unroll
    for (int u = 0; u < 4; ++u) {
      float cs = csum[u] + __shfl_xor(csum[u], 16);
      if (lane < 16) atomicAdd(&colsum[j0 + u * 16 + lane], cs);
    }
  } else {
#pragma unroll
    for (int u = 0; u < 4; ++u) {
      const int j = j0 + u * 16 + col;
      const float cmx = colmax[j];
      float c0 = 0.0f, c1 = 0.0f, c2 = 0.0f;
#pragma unroll
      for (int v = 0; v < 8; ++v) {
        const int i = i0 + v + 8 * hi16;
        float ey = expf(SIGMA_C * (acc[u][v] - cmx));
        c0 += ey * predl[i*3+0]; c1 += ey * predl[i*3+1]; c2 += ey * predl[i*3+2];
      }
      c0 += __shfl_xor(c0, 16); c1 += __shfl_xor(c1, 16); c2 += __shfl_xor(c2, 16);
      if (lane < 16) {
        atomicAdd(&colacc[(j0 + u * 16 + lane)*3+0], c0);
        atomicAdd(&colacc[(j0 + u * 16 + lane)*3+1], c1);
        atomicAdd(&colacc[(j0 + u * 16 + lane)*3+2], c2);
      }
    }
  }
}

__global__ void k_mapped(const float* __restrict__ rowacc, const float* __restrict__ rowsum,
                         float* __restrict__ mapped, int N) {
  int i = blockIdx.x * blockDim.x + threadIdx.x;
  if (i >= N) return;
  float inv = 1.0f / fmaxf(rowsum[i], 1e-30f);
  mapped[i*3+0] = rowacc[i*3+0] * inv;
  mapped[i*3+1] = rowacc[i*3+1] * inv;
  mapped[i*3+2] = rowacc[i*3+2] * inv;
}

// small fc (c_dim=3) fused with pred_t = vx + disp*t
__global__ void k_fc_small_pred(const float* __restrict__ F, int W,
                                const float* __restrict__ Wf, const float* __restrict__ bf,
                                const float* __restrict__ vx, float t,
                                float* __restrict__ pred, int N) {
  int i = blockIdx.x * blockDim.x + threadIdx.x;
  if (i >= N) return;
  float a0 = bf[0], a1 = bf[1], a2 = bf[2];
  const float* r = F + (size_t)i * FWID;
  for (int k = 0; k < W; ++k) {
    float x = r[k];
    a0 += x * Wf[k*3+0]; a1 += x * Wf[k*3+1]; a2 += x * Wf[k*3+2];
  }
  pred[i*3+0] = vx[i*3+0] + a0 * t;
  pred[i*3+1] = vx[i*3+1] + a1 * t;
  pred[i*3+2] = vx[i*3+2] + a2 * t;
}

// ---------------------------------------------------------------------------
// ARAP: H accumulation, polar rotation (Newton on H^T), residual
// ---------------------------------------------------------------------------
__global__ void k_arapH(const float* __restrict__ a, const float* __restrict__ b,
                        const int* __restrict__ src, const int* __restrict__ dst,
                        int E, float* __restrict__ H) {
  int e = blockIdx.x * blockDim.x + threadIdx.x;
  if (e >= E) return;
  int s = src[e], d = dst[e];
  float ea[3], eb[3];
#pragma unroll
  for (int c = 0; c < 3; ++c) { ea[c] = a[s*3+c] - a[d*3+c]; eb[c] = b[s*3+c] - b[d*3+c]; }
#pragma unroll
  for (int r = 0; r < 3; ++r)
#pragma unroll
    for (int c = 0; c < 3; ++c)
      atomicAdd(&H[(size_t)d*9 + r*3 + c], ea[r] * eb[c]);
}

__global__ void k_arapRot(const float* __restrict__ H, float* __restrict__ R, int N) {
  int i = blockIdx.x * blockDim.x + threadIdx.x;
  if (i >= N) return;
  float X[3][3];
#pragma unroll
  for (int r = 0; r < 3; ++r)
#pragma unroll
    for (int c = 0; c < 3; ++c)
      X[r][c] = H[(size_t)i*9 + c*3 + r];      // A = H^T
  X[0][0] += 1e-6f; X[1][1] += 1e-6f; X[2][2] += 1e-6f;
  for (int it = 0; it < 8; ++it) {             // Newton polar: X = 0.5(X + X^{-T})
    float C[3][3];
    C[0][0] =  X[1][1]*X[2][2] - X[1][2]*X[2][1];
    C[0][1] = -(X[1][0]*X[2][2] - X[1][2]*X[2][0]);
    C[0][2] =  X[1][0]*X[2][1] - X[1][1]*X[2][0];
    C[1][0] = -(X[0][1]*X[2][2] - X[0][2]*X[2][1]);
    C[1][1] =  X[0][0]*X[2][2] - X[0][2]*X[2][0];
    C[1][2] = -(X[0][0]*X[2][1] - X[0][1]*X[2][0]);
    C[2][0] =  X[0][1]*X[1][2] - X[0][2]*X[1][1];
    C[2][1] = -(X[0][0]*X[1][2] - X[0][2]*X[1][0]);
    C[2][2] =  X[0][0]*X[1][1] - X[0][1]*X[1][0];
    float det = X[0][0]*C[0][0] + X[0][1]*C[0][1] + X[0][2]*C[0][2];
    if (fabsf(det) < 1e-12f) det = (det < 0.0f ? -1e-12f : 1e-12f);
    float inv = 1.0f / det;
#pragma unroll
    for (int r = 0; r < 3; ++r)
#pragma unroll
      for (int c = 0; c < 3; ++c)
        X[r][c] = 0.5f * (X[r][c] + C[r][c] * inv);   // inv(X)^T = cof(X)/det
  }
#pragma unroll
  for (int r = 0; r < 3; ++r)
#pragma unroll
    for (int c = 0; c < 3; ++c)
      R[(size_t)i*9 + r*3 + c] = X[r][c];
}

__global__ void k_arapRes(const float* __restrict__ a, const float* __restrict__ b,
                          const int* __restrict__ src, const int* __restrict__ dst,
                          int E, const float* __restrict__ R, float* __restrict__ acc) {
  int e = blockIdx.x * blockDim.x + threadIdx.x;
  float s = 0.0f;
  if (e < E) {
    int si = src[e], d = dst[e];
    float ea[3], eb[3];
#pragma unroll
    for (int c = 0; c < 3; ++c) { ea[c] = a[si*3+c] - a[d*3+c]; eb[c] = b[si*3+c] - b[d*3+c]; }
    const float* r = R + (size_t)d * 9;
#pragma unroll
    for (int rr = 0; rr < 3; ++rr) {
      float v = eb[rr] - (r[rr*3+0]*ea[0] + r[rr*3+1]*ea[1] + r[rr*3+2]*ea[2]);
      s += v * v;
    }
  }
  for (int o = 1; o < 32; o <<= 1) s += __shfl_xor(s, o);
  if ((threadIdx.x & 31) == 0) atomicAdd(acc, s);
}

__global__ void k_align1(const float* __restrict__ mapped, const float* __restrict__ predl,
                         float* __restrict__ acc, int N) {
  int i = blockIdx.x * blockDim.x + threadIdx.x;
  float s = 0.0f;
  if (i < N) {
#pragma unroll
    for (int c = 0; c < 3; ++c) { float d = mapped[i*3+c] - predl[i*3+c]; s += d * d; }
  }
  for (int o = 1; o < 32; o <<= 1) s += __shfl_xor(s, o);
  if ((threadIdx.x & 31) == 0) atomicAdd(acc, s);
}

__global__ void k_align2(const float* __restrict__ vy, const float* __restrict__ colacc,
                         const float* __restrict__ colsum, float* __restrict__ acc, int N) {
  int j = blockIdx.x * blockDim.x + threadIdx.x;
  float s = 0.0f;
  if (j < N) {
    float inv = 1.0f / fmaxf(colsum[j], 1e-30f);
#pragma unroll
    for (int c = 0; c < 3; ++c) { float d = vy[j*3+c] - colacc[j*3+c] * inv; s += d * d; }
  }
  for (int o = 1; o < 32; o <<= 1) s += __shfl_xor(s, o);
  if ((threadIdx.x & 31) == 0) atomicAdd(acc, s);
}

__global__ void k_finalize(const float* __restrict__ accs, float* __restrict__ out, int Nx) {
  if (blockIdx.x == 0 && threadIdx.x == 0) {
    float arap = 0.0f;
    for (int i = 0; i < 8; ++i) arap += accs[i];          // LAMBD_ARAP = 1
    float align = (201.0f / (float)Nx) * 10.0f * (accs[8] + accs[9]);
    out[0] = arap + align;
    out[1] = arap;
    out[2] = align;
    out[3] = 0.0f;                                        // E_geo
  }
}

// ---------------------------------------------------------------------------
extern "C" void kernel_launch(void* const* d_in, const int* in_sizes, int n_in,
                              void* d_out, int out_size, void* d_ws, size_t ws_size,
                              hipStream_t stream) {
  const float* verts_x   = (const float*)d_in[0];
  const float* verts_y   = (const float*)d_in[1];
  const float* normals_x = (const float*)d_in[2];
  const float* normals_y = (const float*)d_in[3];
  const int*   ei_x      = (const int*)d_in[4];
  const int*   ei_y      = (const int*)d_in[5];
  const int N = in_sizes[0] / 3;
  const int E = in_sizes[4] / 2;
  const int* src_x = ei_x;     const int* dst_x = ei_x + E;
  const int* src_y = ei_y;     const int* dst_y = ei_y + E;

  // pytree leaf order: dict keys sorted -> "convs" (5x (W1,b1,W2,b2)) then "fc" (Wf,bf)
  const float *rnW1[5], *rnB1[5], *rnW2[5], *rnB2[5];
  const float *ftW1[5], *ftB1[5], *ftW2[5], *ftB2[5];
  for (int k = 0; k < 5; ++k) {
    rnW1[k] = (const float*)d_in[6  + 4*k]; rnB1[k] = (const float*)d_in[7  + 4*k];
    rnW2[k] = (const float*)d_in[8  + 4*k]; rnB2[k] = (const float*)d_in[9  + 4*k];
    ftW1[k] = (const float*)d_in[28 + 4*k]; ftB1[k] = (const float*)d_in[29 + 4*k];
    ftW2[k] = (const float*)d_in[30 + 4*k]; ftB2[k] = (const float*)d_in[31 + 4*k];
  }
  const float* rnWf = (const float*)d_in[26]; const float* rnBf = (const float*)d_in[27];
  const float* ftWf = (const float*)d_in[48]; const float* ftBf = (const float*)d_in[49];

  // ---- workspace carve ----
  char* w8 = (char*)d_ws;
  size_t off = 0;
  auto carve = [&](size_t bytes) -> void* {
    off = (off + 255) & ~(size_t)255;
    void* p = w8 + off; off += bytes; return p;
  };
  float*     Fx     = (float*)carve((size_t)N * FWID * 4);
  float*     Fy     = (float*)carve((size_t)N * FWID * 4);
  float*     Frn    = (float*)carve((size_t)N * FWID * 4);
  _Float16*  H1     = (_Float16*)carve((size_t)E * HID * 2);
  float*     S      = (float*)carve((size_t)N * HID * 4);
  float*     cntx   = (float*)carve((size_t)N * 4);
  float*     cnty   = (float*)carve((size_t)N * 4);
  _Float16*  BT1    = (_Float16*)carve((size_t)128 * 1056 * 2);
  _Float16*  BT2    = (_Float16*)carve((size_t)128 * 128 * 2);
  _Float16*  BTf    = (_Float16*)carve((size_t)128 * 672 * 2);
  float*     fxr    = (float*)carve((size_t)N * HID * 4);
  float*     fyr    = (float*)carve((size_t)N * HID * 4);
  _Float16*  fxh    = (_Float16*)carve((size_t)N * HID * 2);
  _Float16*  fyh    = (_Float16*)carve((size_t)N * HID * 2);
  float*     rowmax = (float*)carve((size_t)N * 4);
  float*     colmax = (float*)carve((size_t)N * 4);
  float*     rowsum = (float*)carve((size_t)N * 4);
  float*     colsum = (float*)carve((size_t)N * 4);
  float*     rowacc = (float*)carve((size_t)N * 3 * 4);
  float*     colacc = (float*)carve((size_t)N * 3 * 4);
  float*     mapped = (float*)carve((size_t)N * 3 * 4);
  float*     pred   = (float*)carve((size_t)4 * N * 3 * 4);
  float*     Hm     = (float*)carve((size_t)N * 9 * 4);
  float*     Rm     = (float*)carve((size_t)N * 9 * 4);
  float*     accs   = (float*)carve(64 * 4);

  auto fill = [&](float* p, float v, size_t n) {
    k_fill<<<dim3((unsigned)((n + 255) / 256)), dim3(256), 0, stream>>>(p, v, (int)n);
  };
  const dim3 B256(256);
  const unsigned gE   = (unsigned)((E + 255) / 256);
  const unsigned gN   = (unsigned)((N + 255) / 256);
  const unsigned gT32 = (unsigned)((E + 31) / 32);
  const unsigned gN32 = (unsigned)((N + 31) / 32);

  auto run_convs = [&](float* F, int d,
                       const float* const* W1a, const float* const* B1a,
                       const float* const* W2a, const float* const* B2a,
                       const int* sp, const int* dp, const float* cnt) {
    for (int k = 0; k < 5; ++k) {
      int w  = k * 128 + d;
      int K1 = 2 * w;
      int Kp = ((K1 + 31) / 32) * 32;
      k_wt_trans<<<dim3((unsigned)((128 * Kp + 255) / 256)), B256, 0, stream>>>(W1a[k], BT1, K1, 128, Kp);
      k_wt_trans<<<dim3((unsigned)((128 * 128 + 255) / 256)), B256, 0, stream>>>(W2a[k], BT2, 128, 128, 128);
      k_ec_gemm1<<<dim3(gT32), B256, 0, stream>>>(F, (5 - k) * 128, w, sp, dp, E, BT1, Kp, B1a[k], H1);
      k_ec_gemm2<<<dim3(gT32), B256, 0, stream>>>(H1, dp, E, BT2, B2a[k], S);
      k_mean_place<<<dim3((unsigned)((N * HID + 255) / 256)), B256, 0, stream>>>(S, cnt, F, (4 - k) * 128, N);
    }
  };

  // ---- init ----
  fill(S, 0.0f, (size_t)N * HID);
  fill(cntx, 0.0f, N); fill(cnty, 0.0f, N);
  fill(accs, 0.0f, 64);
  k_count<<<dim3(gE), B256, 0, stream>>>(dst_x, E, cntx);
  k_count<<<dim3(gE), B256, 0, stream>>>(dst_y, E, cnty);

  // ---- feature nets ----
  k_init_feat<<<dim3(gN), B256, 0, stream>>>(verts_x, normals_x, Fx, N);
  k_init_feat<<<dim3(gN), B256, 0, stream>>>(verts_y, normals_y, Fy, N);
  run_convs(Fx, 6, ftW1, ftB1, ftW2, ftB2, src_x, dst_x, cntx);
  run_convs(Fy, 6, ftW1, ftB1, ftW2, ftB2, src_y, dst_y, cnty);
  k_wt_trans<<<dim3((unsigned)((128 * 672 + 255) / 256)), B256, 0, stream>>>(ftWf, BTf, 646, 128, 672);
  k_fc_gemm<<<dim3(gN32), B256, 0, stream>>>(Fx, 646, N, BTf, 672, ftBf, fxr);
  k_fc_gemm<<<dim3(gN32), B256, 0, stream>>>(Fy, 646, N, BTf, 672, ftBf, fyr);
  k_normalize<<<dim3(gN), B256, 0, stream>>>(fxr, fxh, N);
  k_normalize<<<dim3(gN), B256, 0, stream>>>(fyr, fyh, N);

  // ---- correspondence (streaming softmax over D) ----
  fill(rowmax, -1e30f, N); fill(colmax, -1e30f, N);
  fill(rowsum, 0.0f, N);   fill(colsum, 0.0f, N);
  fill(rowacc, 0.0f, (size_t)N * 3); fill(colacc, 0.0f, (size_t)N * 3);
  const unsigned jBlocks = (unsigned)((N + 63) / 64);
  dim3 dgrid((unsigned)((N + 15) / 16), (jBlocks + 7) / 8);
  k_dpass<<<dgrid, B256, 0, stream>>>(fxh, fyh, N, N, 0, rowmax, colmax, rowsum, colsum,
                                      rowacc, verts_y, colacc, (const float*)nullptr);
  k_dpass<<<dgrid, B256, 0, stream>>>(fxh, fyh, N, N, 1, rowmax, colmax, rowsum, colsum,
                                      rowacc, verts_y, colacc, (const float*)nullptr);
  k_mapped<<<dim3(gN), B256, 0, stream>>>(rowacc, rowsum, mapped, N);

  // ---- deformation net for 4 timesteps ----
  for (int t = 0; t < 4; ++t) {
    float tv = 0.25f * (float)(t + 1);
    k_init_rn<<<dim3(gN), B256, 0, stream>>>(verts_x, mapped, tv, Frn, N);
    run_convs(Frn, 7, rnW1, rnB1, rnW2, rnB2, src_x, dst_x, cntx);
    k_fc_small_pred<<<dim3(gN), B256, 0, stream>>>(Frn, 647, rnWf, rnBf, verts_x, tv,
                                                   pred + (size_t)t * N * 3, N);
  }
  const float* P0 = pred;
  const float* P1 = pred + (size_t)N * 3;
  const float* P2 = pred + (size_t)2 * N * 3;
  const float* P3 = pred + (size_t)3 * N * 3;

  // ---- Pi_inv @ pred_last + alignment ----
  k_dpass<<<dgrid, B256, 0, stream>>>(fxh, fyh, N, N, 2, rowmax, colmax, rowsum, colsum,
                                      rowacc, verts_y, colacc, P3);
  k_align1<<<dim3(gN), B256, 0, stream>>>(mapped, P3, accs + 8, N);
  k_align2<<<dim3(gN), B256, 0, stream>>>(verts_y, colacc, colsum, accs + 9, N);

  // ---- ARAP terms (8 invocations) ----
  auto arap = [&](const float* a, const float* b, float* accp) {
    fill(Hm, 0.0f, (size_t)N * 9);
    k_arapH<<<dim3(gE), B256, 0, stream>>>(a, b, src_x, dst_x, E, Hm);
    k_arapRot<<<dim3(gN), B256, 0, stream>>>(Hm, Rm, N);
    k_arapRes<<<dim3(gE), B256, 0, stream>>>(a, b, src_x, dst_x, E, Rm, accp);
  };
  arap(verts_x, P0, accs + 0); arap(P0, verts_x, accs + 1);
  arap(P0, P1, accs + 2);      arap(P1, P0, accs + 3);
  arap(P1, P2, accs + 4);      arap(P2, P1, accs + 5);
  arap(P2, P3, accs + 6);      arap(P3, P2, accs + 7);

  k_finalize<<<dim3(1), dim3(1), 0, stream>>>(accs, (float*)d_out, N);
  (void)n_in; (void)out_size; (void)ws_size; (void)in_sizes;
}
